// MultiHeadSelfAttention_36636071035087
// MI455X (gfx1250) — compile-verified, matrix operand order checked by reading, not run on hardware
//
#include <hip/hip_runtime.h>
#include <hip/hip_bf16.h>

// ---------------- problem constants ----------------
#define S_   4
#define B_   2048
#define D_   256
#define H_   8
#define DK_  32
#define SB_  (S_ * B_)        // 8192 rows
#define KTOP 204              // max(1, int(2048*0.1))
#define LN_EPS 1e-5f

typedef __attribute__((ext_vector_type(16))) __bf16 v16bf;
typedef __attribute__((ext_vector_type(8)))  float  v8f;
typedef __attribute__((ext_vector_type(4)))  unsigned int v4u;

static __device__ inline v8f zero8() {
  v8f z;
#pragma unroll
  for (int i = 0; i < 8; ++i) z[i] = 0.0f;
  return z;
}

// Load a 16-half WMMA fragment: halves 0..7 from p0 (16B), 8..15 from p1 (16B).
static __device__ inline v16bf load_frag(const __hip_bfloat16* p0,
                                         const __hip_bfloat16* p1) {
  union { v16bf bf; v4u u[2]; } f;
  f.u[0] = *(const v4u*)p0;
  f.u[1] = *(const v4u*)p1;
  return f.bf;
}

static __device__ inline v8f wmma_bf16(v16bf a, v16bf b, v8f c) {
  return __builtin_amdgcn_wmma_f32_16x16x32_bf16(false, a, false, b,
                                                 (short)0, c, false, false);
}

// ---------------- weight f32 -> bf16 ----------------
__global__ void cvt_w_kernel(const float* wq, const float* wk, const float* wv,
                             const float* wo, __hip_bfloat16* wqb,
                             __hip_bfloat16* wkb, __hip_bfloat16* wvb,
                             __hip_bfloat16* wob) {
  int i = blockIdx.x * 256 + threadIdx.x;      // 0 .. 4*65536-1
  int m = i >> 16;
  int r = i & 65535;
  const float* src = (m == 0) ? wq : (m == 1) ? wk : (m == 2) ? wv : wo;
  __hip_bfloat16* dst = (m == 0) ? wqb : (m == 1) ? wkb : (m == 2) ? wvb : wob;
  dst[r] = __float2bfloat16(src[r]);
}

// ---------------- LayerNorm (one block per row) ----------------
__global__ void ln_kernel(const float* __restrict__ x,
                          const float* __restrict__ lnw,
                          const float* __restrict__ lnb,
                          __hip_bfloat16* __restrict__ xn) {
  __shared__ float red[256];
  int row = blockIdx.x;
  int t = threadIdx.x;
  float v = x[row * D_ + t];
  red[t] = v;
  __syncthreads();
#pragma unroll
  for (int s = 128; s > 0; s >>= 1) {
    if (t < s) red[t] += red[t + s];
    __syncthreads();
  }
  float mu = red[0] * (1.0f / D_);
  __syncthreads();
  float d = v - mu;
  red[t] = d * d;
  __syncthreads();
#pragma unroll
  for (int s = 128; s > 0; s >>= 1) {
    if (t < s) red[t] += red[t + s];
    __syncthreads();
  }
  float var = red[0] * (1.0f / D_);
  float out = d * rsqrtf(var + LN_EPS) * lnw[t] + lnb[t];
  xn[row * D_ + t] = __float2bfloat16(out);
}

// ---------------- QKV projection GEMM ----------------
// One wave computes one 16x16 tile of (xn @ w.T).  q,k stored [s,h,b,dk];
// v stored transposed [s,h,dk,b] so it is directly a WMMA B-fragment later.
__global__ void qkv_kernel(const __hip_bfloat16* __restrict__ xn,
                           const __hip_bfloat16* __restrict__ wqb,
                           const __hip_bfloat16* __restrict__ wkb,
                           const __hip_bfloat16* __restrict__ wvb,
                           __hip_bfloat16* __restrict__ qb,
                           __hip_bfloat16* __restrict__ kb,
                           __hip_bfloat16* __restrict__ vtb) {
  int wave = threadIdx.x >> 5;
  int lane = threadIdx.x & 31;
  int mtile = blockIdx.x;                 // 0..511
  int ntile = blockIdx.y * 8 + wave;      // 0..15
  int mat = blockIdx.z;                   // 0=q 1=k 2=v
  const __hip_bfloat16* w = (mat == 0) ? wqb : (mat == 1) ? wkb : wvb;

  int row = mtile * 16 + (lane & 15);
  int n = ntile * 16 + (lane & 15);
  int kb8 = (lane >> 4) * 8;

  v8f c = zero8();
#pragma unroll
  for (int kc = 0; kc < 8; ++kc) {
    int kbase = kc * 32 + kb8;
    v16bf a = load_frag(&xn[row * D_ + kbase], &xn[row * D_ + kbase + 16]);
    v16bf b = load_frag(&w[n * D_ + kbase], &w[n * D_ + kbase + 16]);
    c = wmma_bf16(a, b, c);
  }

  int mloc = (lane >> 4) << 3;
  int h = n >> 5, dk = n & 31;
#pragma unroll
  for (int j = 0; j < 8; ++j) {
    int m = mtile * 16 + j + mloc;
    int s = m >> 11;            // /B_
    int bidx = m & (B_ - 1);
    __hip_bfloat16 val = __float2bfloat16(c[j]);
    if (mat == 0)
      qb[(((s * H_ + h) * B_) + bidx) * DK_ + dk] = val;
    else if (mat == 1)
      kb[(((s * H_ + h) * B_) + bidx) * DK_ + dk] = val;
    else
      vtb[(((s * H_ + h) * DK_) + dk) * B_ + bidx] = val;
  }
}

// ---------------- attention: scores -> topk threshold -> softmax -> ctx ----
// grid.x = S*H*128, block = 256 (8 waves).  Dynamic LDS:
//   sc[16][2048] f32 scores (reused as ctx staging), pred[256], ccnt[16],
//   rmax[16], rmin[16]
__global__ void attn_kernel(const __hip_bfloat16* __restrict__ qb,
                            const __hip_bfloat16* __restrict__ kb,
                            const __hip_bfloat16* __restrict__ vtb,
                            __hip_bfloat16* __restrict__ ctxb) {
  extern __shared__ float smem[];
  float* sc = smem;                     // 16*2048
  float* pred = sc + 16 * B_;           // 256
  float* ccnt = pred + 256;             // 16
  float* rmax = ccnt + 16;              // 16
  float* rmin = rmax + 16;              // 16

  int t = threadIdx.x;
  int wave = t >> 5;
  int lane = t & 31;
  int rt = blockIdx.x & 127;            // row tile within head
  int sh = blockIdx.x >> 7;             // s*H + h
  const __hip_bfloat16* qh = qb + (size_t)sh * B_ * DK_;
  const __hip_bfloat16* kh = kb + (size_t)sh * B_ * DK_;
  const __hip_bfloat16* vh = vtb + (size_t)sh * DK_ * B_;

  int kb8 = (lane >> 4) * 8;
  int mloc = (lane >> 4) << 3;

  // ---- phase 1: scores for this wave's 256-key segment (K = DK = 32) ----
  {
    int row = rt * 16 + (lane & 15);
    v16bf aq = load_frag(&qh[row * DK_ + kb8], &qh[row * DK_ + kb8 + 16]);
    const float scl = 0.17677669529663687f;  // 1/sqrt(32)
#pragma unroll
    for (int kt = 0; kt < 16; ++kt) {
      int key = wave * 256 + kt * 16 + (lane & 15);
      v16bf bk = load_frag(&kh[key * DK_ + kb8], &kh[key * DK_ + kb8 + 16]);
      v8f c = zero8();
      c = wmma_bf16(aq, bk, c);
#pragma unroll
      for (int j = 0; j < 8; ++j)
        sc[(j + mloc) * B_ + key] = c[j] * scl;
    }
  }
  __syncthreads();

  // ---- phase 2: per-row top-k threshold via value bisection ----
  int r = t >> 4;        // row 0..15
  int seg = t & 15;      // 128-element chunk
  int base = r * B_ + seg * 128;
  float mx = -1e30f, mn = 1e30f;
  for (int i = 0; i < 128; ++i) {
    float v = sc[base + i];
    mx = fmaxf(mx, v);
    mn = fminf(mn, v);
  }
  pred[t] = mx;
  __syncthreads();
  if (t < 16) {
    float m = -1e30f;
    for (int j = 0; j < 16; ++j) m = fmaxf(m, pred[t * 16 + j]);
    rmax[t] = m;
  }
  __syncthreads();
  pred[t] = mn;
  __syncthreads();
  if (t < 16) {
    float m = 1e30f;
    for (int j = 0; j < 16; ++j) m = fminf(m, pred[t * 16 + j]);
    rmin[t] = m;
  }
  __syncthreads();

  float lo = rmin[r] - 1.0f;   // count(>=lo) = 2048 >= KTOP
  float hi = rmax[r] + 1.0f;   // count(>=hi) = 0    <  KTOP
  for (int it = 0; it < 26; ++it) {
    float mid = 0.5f * (lo + hi);
    float cnt = 0.0f;
    for (int i = 0; i < 128; ++i)
      cnt += (sc[base + i] >= mid) ? 1.0f : 0.0f;
    pred[t] = cnt;
    __syncthreads();
    if (t < 16) {
      float tot = 0.0f;
      for (int j = 0; j < 16; ++j) tot += pred[t * 16 + j];
      ccnt[t] = tot;
    }
    __syncthreads();
    if (ccnt[r] >= (float)KTOP) lo = mid; else hi = mid;
    __syncthreads();
  }
  float thr = lo;

  // ---- phase 3: masked exp + row sum (normalization folded in later) ----
  float rm = rmax[r];
  float psum = 0.0f;
  for (int i = 0; i < 128; ++i) {
    float v = sc[base + i];
    float p = (v >= thr) ? __expf(v - rm) : 0.0f;
    sc[base + i] = p;
    psum += p;
  }
  pred[t] = psum;
  __syncthreads();
  if (t < 16) {
    float tot = 0.0f;
    for (int j = 0; j < 16; ++j) tot += pred[t * 16 + j];
    ccnt[t] = 1.0f / tot;     // reuse ccnt as inv row sum
  }
  __syncthreads();

  // ---- phase 4: ctx partial = attn_seg @ V_seg (16x256 @ 256x32) ----
  v8f a0 = zero8(), a1 = zero8();
  {
    int rowA = lane & 15;
#pragma unroll
    for (int kc = 0; kc < 8; ++kc) {
      int key0 = wave * 256 + kc * 32;
      v16bf af;
#pragma unroll
      for (int e = 0; e < 8; ++e)
        af[e] = (__bf16)sc[rowA * B_ + key0 + kb8 + e];
#pragma unroll
      for (int e = 0; e < 8; ++e)
        af[8 + e] = (__bf16)sc[rowA * B_ + key0 + kb8 + 16 + e];
      int n0 = lane & 15;
      v16bf b0 = load_frag(&vh[n0 * B_ + key0 + kb8],
                           &vh[n0 * B_ + key0 + kb8 + 16]);
      v16bf b1 = load_frag(&vh[(16 + n0) * B_ + key0 + kb8],
                           &vh[(16 + n0) * B_ + key0 + kb8 + 16]);
      a0 = wmma_bf16(af, b0, a0);
      a1 = wmma_bf16(af, b1, a1);
    }
  }
  __syncthreads();   // everyone done reading sc -> safe to reuse as staging

  // stage per-wave 16x32 partials into sc[wave*512 ...]
#pragma unroll
  for (int j = 0; j < 8; ++j) {
    int ml = j + mloc;
    sc[wave * 512 + ml * 32 + (lane & 15)] = a0[j];
    sc[wave * 512 + ml * 32 + 16 + (lane & 15)] = a1[j];
  }
  __syncthreads();

  // reduce over 8 waves, scale by inv row-sum, store ctx bf16 in (S,B,D)
  int sidx = sh >> 3, h = sh & 7;
  for (int e = t; e < 512; e += 256) {
    float acc = 0.0f;
#pragma unroll
    for (int w2 = 0; w2 < 8; ++w2) acc += sc[w2 * 512 + e];
    int ml = e >> 5, dk = e & 31;
    float val = acc * ccnt[ml];
    int bgl = rt * 16 + ml;
    ctxb[((size_t)(sidx * B_ + bgl)) * D_ + h * DK_ + dk] =
        __float2bfloat16(val);
  }
}

// ---------------- output projection + bias + residual ----------------
__global__ void proj_kernel(const __hip_bfloat16* __restrict__ ctxb,
                            const __hip_bfloat16* __restrict__ wob,
                            const float* __restrict__ wo_bias,
                            const float* __restrict__ x,
                            float* __restrict__ out) {
  int wave = threadIdx.x >> 5;
  int lane = threadIdx.x & 31;
  int mtile = blockIdx.x;
  int ntile = blockIdx.y * 8 + wave;

  int row = mtile * 16 + (lane & 15);
  int n = ntile * 16 + (lane & 15);
  int kb8 = (lane >> 4) * 8;

  v8f c = zero8();
#pragma unroll
  for (int kc = 0; kc < 8; ++kc) {
    int kbase = kc * 32 + kb8;
    v16bf a = load_frag(&ctxb[row * D_ + kbase], &ctxb[row * D_ + kbase + 16]);
    v16bf b = load_frag(&wob[n * D_ + kbase], &wob[n * D_ + kbase + 16]);
    c = wmma_bf16(a, b, c);
  }

  int mloc = (lane >> 4) << 3;
  float bias = wo_bias[n];
#pragma unroll
  for (int j = 0; j < 8; ++j) {
    int m = mtile * 16 + j + mloc;
    out[m * D_ + n] = c[j] + bias + x[m * D_ + n];
  }
}

// ---------------- launch ----------------
extern "C" void kernel_launch(void* const* d_in, const int* in_sizes, int n_in,
                              void* d_out, int out_size, void* d_ws,
                              size_t ws_size, hipStream_t stream) {
  const float* x   = (const float*)d_in[0];
  const float* wq  = (const float*)d_in[1];
  const float* wk  = (const float*)d_in[2];
  const float* wv  = (const float*)d_in[3];
  const float* wo  = (const float*)d_in[4];
  const float* wob_f = (const float*)d_in[5];
  const float* lnw = (const float*)d_in[6];
  const float* lnb = (const float*)d_in[7];

  char* ws = (char*)d_ws;
  const size_t MB = 1024 * 1024;
  __hip_bfloat16* xn  = (__hip_bfloat16*)(ws + 0 * MB);   // 8192*256 bf16 = 4MB
  __hip_bfloat16* qb  = (__hip_bfloat16*)(ws + 4 * MB);   // 4MB
  __hip_bfloat16* kb  = (__hip_bfloat16*)(ws + 8 * MB);   // 4MB
  __hip_bfloat16* vtb = (__hip_bfloat16*)(ws + 12 * MB);  // 4MB (transposed V)
  __hip_bfloat16* ctx = (__hip_bfloat16*)(ws + 16 * MB);  // 4MB
  __hip_bfloat16* wqb = (__hip_bfloat16*)(ws + 20 * MB);
  __hip_bfloat16* wkb = (__hip_bfloat16*)(ws + 20 * MB + 128 * 1024);
  __hip_bfloat16* wvb = (__hip_bfloat16*)(ws + 20 * MB + 256 * 1024);
  __hip_bfloat16* wob = (__hip_bfloat16*)(ws + 20 * MB + 384 * 1024);

  cvt_w_kernel<<<1024, 256, 0, stream>>>(wq, wk, wv, wo, wqb, wkb, wvb, wob);
  ln_kernel<<<SB_, 256, 0, stream>>>(x, lnw, lnb, xn);
  qkv_kernel<<<dim3(SB_ / 16, 2, 3), 256, 0, stream>>>(xn, wqb, wkb, wvb, qb,
                                                       kb, vtb);
  size_t shmem = (size_t)(16 * B_ + 256 + 16 + 16 + 16) * sizeof(float);
  attn_kernel<<<S_ * H_ * (B_ / 16), 256, shmem, stream>>>(qb, kb, vtb, ctx);
  proj_kernel<<<dim3(SB_ / 16, 2), 256, 0, stream>>>(ctx, wob, wob_f, x,
                                                     (float*)d_out);
}
